// CNNGRUv1_26379689132566
// MI455X (gfx1250) — matrix operand order repaired and checked by reading, not compiled, not run on hardware
//
#include <hip/hip_runtime.h>

// Problem constants (from reference)
#define S_LEN 256
#define B_LEN 512
#define I_LEN 46
#define H_LEN 256
#define C_LEN 40
#define NROWS (S_LEN * B_LEN) // 131072
#define EPSV 1e-5f

#define C1_BLOCKS (NROWS / 32) // 4096
#define C2_BLOCKS (NROWS / 64) // 2048

typedef _Float16 h16;
typedef __attribute__((ext_vector_type(16))) _Float16 v16h;
typedef __attribute__((ext_vector_type(8)))  _Float16 v8h;
typedef __attribute__((ext_vector_type(8)))  float    v8f;

// ---------------------------------------------------------------------------
// WMMA fragment helpers (CDNA5 ISA 7.12.2 layouts, wave32)
// A (16x32 f16): lane L -> row m = mbase + (L&15); elements 0..7 = K[k0 + 8*(L>>4) ..],
//                elements 8..15 = K[k0 + 16 + 8*(L>>4) ..]  (two contiguous 8-half groups)
// B (32x16 f16): lane L -> col n = nbase + (L&15); same K grouping as A
// C/D (16x16 f32): vgpr r, lane L -> m = r + 8*(L>>4), n = L&15
// ---------------------------------------------------------------------------
__device__ __forceinline__ v16h frag_a_lds(const h16* base, int ldk, int mbase,
                                           int k0, int lane) {
  const int m  = mbase + (lane & 15);
  const int hi = lane >> 4;
  const h16* p = base + m * ldk + k0 + hi * 8;
  v8h lo = *(const v8h*)p;
  v8h hv = *(const v8h*)(p + 16);
  v16h r;
#pragma unroll
  for (int e = 0; e < 8; ++e) { r[e] = lo[e]; r[e + 8] = hv[e]; }
  return r;
}

__device__ __forceinline__ v16h frag_b_glb(const h16* __restrict__ W, int ldk,
                                           int nbase, int koff, int lane) {
  const int n  = lane & 15;
  const int hi = lane >> 4;
  const h16* p = W + (size_t)(nbase + n) * ldk + koff + hi * 8;
  v8h lo = *(const v8h*)p;
  v8h hv = *(const v8h*)(p + 16);
  v16h r;
#pragma unroll
  for (int e = 0; e < 8; ++e) { r[e] = lo[e]; r[e + 8] = hv[e]; }
  return r;
}

__device__ __forceinline__ v8f wmma16(v16h a, v16h b, v8f c) {
  // D = A x B + C, f32 accumulate  ->  v_wmma_f32_16x16x32_f16
  return __builtin_amdgcn_wmma_f32_16x16x32_f16(false, a, false, b, (short)0, c,
                                                false, false);
}

__device__ __forceinline__ float sigm(float x) { return 1.f / (1.f + __expf(-x)); }

// ---------------------------------------------------------------------------
// Utility kernels
// ---------------------------------------------------------------------------
__global__ void k_cvt_f32_f16(const float* __restrict__ src, h16* __restrict__ dst, int n) {
  for (int i = blockIdx.x * blockDim.x + threadIdx.x; i < n; i += gridDim.x * blockDim.x)
    dst[i] = (h16)src[i];
}

// h0 = p_info @ i2h_w.T + i2h_b ; p_info = X[0, b, 40:46]
__global__ void k_h0(const float* __restrict__ X, const float* __restrict__ w,
                     const float* __restrict__ b, float* __restrict__ h0) {
  const int bb = blockIdx.x, h = threadIdx.x;
  const float* p = X + (size_t)bb * I_LEN + C_LEN;
  float acc = b[h];
#pragma unroll
  for (int k = 0; k < 6; ++k) acc += p[k] * w[h * 6 + k];
  h0[(size_t)bb * H_LEN + h] = acc;
}

// ---------------------------------------------------------------------------
// conv1 path: Xc = [Xin|Min]@comb_w.T + comb_b ; t1 = Xc@conv1_w.T + conv1_b
// Per-block per-channel BN partials written to fixed slots (deterministic).
// ---------------------------------------------------------------------------
__global__ void __launch_bounds__(256) k_conv1(
    const float* __restrict__ X, const float* __restrict__ M,
    const float* __restrict__ comb_w, const float* __restrict__ comb_b,
    const float* __restrict__ conv1_w, const float* __restrict__ conv1_b,
    float* __restrict__ t1, float* __restrict__ part) {
  __shared__ float sXin[32 * C_LEN], sMin[32 * C_LEN], sXc[32 * C_LEN];
  const int tid = threadIdx.x;
  const int rowBase = blockIdx.x * 32;
  for (int idx = tid; idx < 32 * C_LEN; idx += 256) {
    int r = idx / C_LEN, c = idx % C_LEN;
    size_t g = (size_t)(rowBase + r) * I_LEN + c;
    sXin[idx] = X[g];
    sMin[idx] = M[g];
  }
  __syncthreads();
  for (int idx = tid; idx < 32 * C_LEN; idx += 256) {
    int r = idx / C_LEN, c = idx % C_LEN;
    float acc = comb_b[c];
    const float* wr = comb_w + c * (2 * C_LEN);
#pragma unroll 8
    for (int k = 0; k < C_LEN; ++k)
      acc += sXin[r * C_LEN + k] * wr[k] + sMin[r * C_LEN + k] * wr[C_LEN + k];
    sXc[idx] = acc;
  }
  __syncthreads();
  const int ch = tid; // thread == output channel: owns its partials outright
  float wv[C_LEN];
#pragma unroll
  for (int k = 0; k < C_LEN; ++k) wv[k] = conv1_w[ch * C_LEN + k];
  const float bias = conv1_b[ch];
  float lsum = 0.f, lssq = 0.f;
  for (int r = 0; r < 32; ++r) {
    float acc = bias;
#pragma unroll
    for (int k = 0; k < C_LEN; ++k) acc += sXc[r * C_LEN + k] * wv[k];
    t1[(size_t)(rowBase + r) * H_LEN + ch] = acc;
    lsum += acc;
    lssq += acc * acc;
  }
  part[(size_t)blockIdx.x * 512 + ch] = lsum;
  part[(size_t)blockIdx.x * 512 + 256 + ch] = lssq;
}

// Deterministic partial reduction + fold BN stats into scale/shift
__global__ void k_bnredfin(const float* __restrict__ part, int nblocks,
                           const float* __restrict__ g, const float* __restrict__ b,
                           float* __restrict__ scale, float* __restrict__ shift) {
  const int t = threadIdx.x;
  float sum = 0.f, ssq = 0.f;
  for (int i = 0; i < nblocks; ++i) {
    sum += part[(size_t)i * 512 + t];
    ssq += part[(size_t)i * 512 + 256 + t];
  }
  const float inv_n = 1.f / (float)NROWS;
  const float mean = sum * inv_n;
  const float var = ssq * inv_n - mean * mean;
  const float sc = g[t] * rsqrtf(var + EPSV);
  scale[t] = sc;
  shift[t] = b[t] - mean * sc;
}

// ---------------------------------------------------------------------------
// conv2: t2 = relu(bn1(t1)) @ conv2_w.T + conv2_b   (WMMA, K=256)
// 128 threads (4 waves), 64 rows/WG.  Deterministic bn2 partials.
// ---------------------------------------------------------------------------
__global__ void __launch_bounds__(128) k_conv2(
    const float* __restrict__ t1, const float* __restrict__ sc1,
    const float* __restrict__ sh1, const h16* __restrict__ Wf16,
    const float* __restrict__ bias, float* __restrict__ t2,
    float* __restrict__ part) {
  __shared__ h16 abuf[64 * H_LEN]; // 32 KB
  __shared__ float wsum[4 * H_LEN], wssq[4 * H_LEN]; // per-wave partials (8 KB)
  const int tid = threadIdx.x, lane = tid & 31, wave = tid >> 5;
  const int rowBase = blockIdx.x * 64;
  for (int idx = tid; idx < (64 * H_LEN) / 4; idx += 128) {
    const int fi = idx * 4, r = fi >> 8, c = fi & 255;
    const float4 v = *(const float4*)(t1 + (size_t)(rowBase + r) * H_LEN + c);
    abuf[r * H_LEN + c + 0] = (h16)fmaxf(v.x * sc1[c + 0] + sh1[c + 0], 0.f);
    abuf[r * H_LEN + c + 1] = (h16)fmaxf(v.y * sc1[c + 1] + sh1[c + 1], 0.f);
    abuf[r * H_LEN + c + 2] = (h16)fmaxf(v.z * sc1[c + 2] + sh1[c + 2], 0.f);
    abuf[r * H_LEN + c + 3] = (h16)fmaxf(v.w * sc1[c + 3] + sh1[c + 3], 0.f);
  }
  __syncthreads();
  const int mbase = wave * 16;
  const int n = lane & 15, hi = lane >> 4;
  for (int jt = 0; jt < 16; ++jt) {
    v8f acc0 = {}, acc1 = {}; // two chains -> more WMMA ILP
#pragma unroll
    for (int kk = 0; kk < 8; kk += 2) {
      acc0 = wmma16(frag_a_lds(abuf, H_LEN, mbase, kk * 32, lane),
                    frag_b_glb(Wf16, H_LEN, jt * 16, kk * 32, lane), acc0);
      acc1 = wmma16(frag_a_lds(abuf, H_LEN, mbase, kk * 32 + 32, lane),
                    frag_b_glb(Wf16, H_LEN, jt * 16, kk * 32 + 32, lane), acc1);
    }
    const int j = jt * 16 + n;
    const float bj = bias[j];
    float lsum = 0.f, lssq = 0.f;
#pragma unroll
    for (int i = 0; i < 8; ++i) {
      const int m = mbase + i + 8 * hi;
      const float v = acc0[i] + acc1[i] + bj;
      t2[(size_t)(rowBase + m) * H_LEN + j] = v;
      lsum += v;
      lssq += v * v;
    }
    // combine hi halves (deterministic: fixed pairing), lane hi==0 writes
    lsum += __shfl_xor(lsum, 16, 32);
    lssq += __shfl_xor(lssq, 16, 32);
    if (hi == 0) {
      wsum[wave * H_LEN + j] = lsum;
      wssq[wave * H_LEN + j] = lssq;
    }
  }
  __syncthreads();
  for (int j = tid; j < H_LEN; j += 128) {
    const float s = wsum[j] + wsum[H_LEN + j] + wsum[2 * H_LEN + j] + wsum[3 * H_LEN + j];
    const float q = wssq[j] + wssq[H_LEN + j] + wssq[2 * H_LEN + j] + wssq[3 * H_LEN + j];
    part[(size_t)blockIdx.x * 512 + j] = s;
    part[(size_t)blockIdx.x * 512 + 256 + j] = q;
  }
}

// hx = bn2(t2) + Xin@down_w.T + down_b   (residual recomputed, K=40)
__global__ void __launch_bounds__(256) k_hx(
    const float* __restrict__ X, const float* __restrict__ t2,
    const float* __restrict__ sc2, const float* __restrict__ sh2,
    const float* __restrict__ down_w, const float* __restrict__ down_b,
    float* __restrict__ hx) {
  __shared__ float sx[C_LEN];
  const size_t row = blockIdx.x;
  const int h = threadIdx.x;
  if (h < C_LEN) sx[h] = X[row * I_LEN + h];
  __syncthreads();
  float acc = down_b[h];
  const float* wr = down_w + h * C_LEN;
#pragma unroll 8
  for (int k = 0; k < C_LEN; ++k) acc += sx[k] * wr[k];
  hx[row * H_LEN + h] = t2[row * H_LEN + h] * sc2[h] + sh2[h] + acc;
}

// ---------------------------------------------------------------------------
// Persistent GRU kernel: 16 WGs x 32 batch rows, 8 waves/WG (2 per SIMD32),
// loops all S steps.  Per step (all K=256, WMMA f16):
//   phase1: 6 GEMM chains (x@wih_{r,z,n}, h@whh_{r,z,n}) -> gates in-register
//           -> h_new to double-buffered LDS (f16)
//   phase2: out[s] = x@cw[:, :H].T + h_new@cw[:, H:].T + cb
// Biases cached in LDS once; weights stream from L2 (all f16 weights ~2 MB).
// ---------------------------------------------------------------------------
__global__ void __launch_bounds__(256) k_gru(
    const float* __restrict__ xin, const float* __restrict__ h0,
    const float* __restrict__ mask, const h16* __restrict__ wih,
    const h16* __restrict__ whh, const h16* __restrict__ cw,
    const float* __restrict__ bih, const float* __restrict__ bhh,
    const float* __restrict__ cb, float* __restrict__ outs) {
  __shared__ h16 xbuf[32 * H_LEN];     // 16 KB
  __shared__ h16 hbuf[2][32 * H_LEN];  // 32 KB (double-buffered hidden state)
  __shared__ float sbih[3 * H_LEN], sbhh[3 * H_LEN], scb[H_LEN]; // 7 KB
  __shared__ float mbuf[32];
  const int tid = threadIdx.x, lane = tid & 31, wave = tid >> 5;
  const int rowBase = blockIdx.x * 32;
  const int n = lane & 15, hi = lane >> 4;

  // preload biases once
  for (int i = tid; i < 3 * H_LEN; i += 256) { sbih[i] = bih[i]; sbhh[i] = bhh[i]; }
  for (int i = tid; i < H_LEN; i += 256) scb[i] = cb[i];

  // stage h0 -> hbuf[0]
  for (int idx = tid; idx < (32 * H_LEN) / 4; idx += 256) {
    const int fi = idx * 4, m = fi >> 8, c = fi & 255;
    const float4 v = *(const float4*)(h0 + (size_t)(rowBase + m) * H_LEN + c);
    hbuf[0][m * H_LEN + c + 0] = (h16)v.x;
    hbuf[0][m * H_LEN + c + 1] = (h16)v.y;
    hbuf[0][m * H_LEN + c + 2] = (h16)v.z;
    hbuf[0][m * H_LEN + c + 3] = (h16)v.w;
  }

  for (int s = 0; s < S_LEN; ++s) {
    const int cur = s & 1, nxt = cur ^ 1;
    // stage x_s and mask_s
    for (int idx = tid; idx < (32 * H_LEN) / 4; idx += 256) {
      const int fi = idx * 4, m = fi >> 8, c = fi & 255;
      const float4 v =
          *(const float4*)(xin + ((size_t)s * B_LEN + rowBase + m) * H_LEN + c);
      xbuf[m * H_LEN + c + 0] = (h16)v.x;
      xbuf[m * H_LEN + c + 1] = (h16)v.y;
      xbuf[m * H_LEN + c + 2] = (h16)v.z;
      xbuf[m * H_LEN + c + 3] = (h16)v.w;
    }
    if (tid < 32) mbuf[tid] = mask[(size_t)s * B_LEN + rowBase + tid];
    __syncthreads();

    // phase 1: gates + h update (each wave owns 4 (mt,jt) output tiles)
    for (int p = 0; p < 4; ++p) {
      const int q = wave * 4 + p;
      const int mt = q >> 4, jt = q & 15;
      const int mbase = mt * 16;
      v8f axr = {}, axz = {}, axn = {}, ahr = {}, ahz = {}, ahn = {};
#pragma unroll
      for (int kk = 0; kk < 8; ++kk) {
        const int k0 = kk * 32;
        v16h ax = frag_a_lds(xbuf, H_LEN, mbase, k0, lane);
        v16h ah = frag_a_lds(hbuf[cur], H_LEN, mbase, k0, lane);
        axr = wmma16(ax, frag_b_glb(wih, H_LEN, 0 * H_LEN + jt * 16, k0, lane), axr);
        axz = wmma16(ax, frag_b_glb(wih, H_LEN, 1 * H_LEN + jt * 16, k0, lane), axz);
        axn = wmma16(ax, frag_b_glb(wih, H_LEN, 2 * H_LEN + jt * 16, k0, lane), axn);
        ahr = wmma16(ah, frag_b_glb(whh, H_LEN, 0 * H_LEN + jt * 16, k0, lane), ahr);
        ahz = wmma16(ah, frag_b_glb(whh, H_LEN, 1 * H_LEN + jt * 16, k0, lane), ahz);
        ahn = wmma16(ah, frag_b_glb(whh, H_LEN, 2 * H_LEN + jt * 16, k0, lane), ahn);
      }
      const int j = jt * 16 + n;
      const float bir = sbih[j], biz = sbih[j + H_LEN], bin_ = sbih[j + 2 * H_LEN];
      const float bhr = sbhh[j], bhz = sbhh[j + H_LEN], bhn = sbhh[j + 2 * H_LEN];
#pragma unroll
      for (int i = 0; i < 8; ++i) {
        const int m = mbase + i + 8 * hi;
        const float hp = (float)hbuf[cur][m * H_LEN + j];
        const float r = sigm(axr[i] + bir + ahr[i] + bhr);
        const float z = sigm(axz[i] + biz + ahz[i] + bhz);
        const float ng = tanhf(axn[i] + bin_ + r * (ahn[i] + bhn));
        const float hn2 = (1.f - z) * ng + z * hp;
        const float mm = mbuf[m];
        hbuf[nxt][m * H_LEN + j] = (h16)(hn2 * mm + hp * (1.f - mm));
      }
    }
    __syncthreads();

    // phase 2: out[s] = x@cwx + h_new@cwh + cb  (two independent chains)
    for (int p = 0; p < 4; ++p) {
      const int q = wave * 4 + p;
      const int mt = q >> 4, jt = q & 15;
      const int mbase = mt * 16;
      v8f ao1 = {}, ao2 = {};
#pragma unroll
      for (int kk = 0; kk < 8; ++kk) {
        const int k0 = kk * 32;
        ao1 = wmma16(frag_a_lds(xbuf, H_LEN, mbase, k0, lane),
                     frag_b_glb(cw, 2 * H_LEN, jt * 16, k0, lane), ao1);
        ao2 = wmma16(frag_a_lds(hbuf[nxt], H_LEN, mbase, k0, lane),
                     frag_b_glb(cw, 2 * H_LEN, jt * 16, H_LEN + k0, lane), ao2);
      }
      const int j = jt * 16 + n;
      const float cbv = scb[j];
#pragma unroll
      for (int i = 0; i < 8; ++i) {
        const int m = mbase + i + 8 * hi;
        outs[((size_t)s * B_LEN + rowBase + m) * H_LEN + j] = ao1[i] + ao2[i] + cbv;
      }
    }
    __syncthreads();
  }
}

// LayerNorm over H=256, wave32 per row, shfl_xor reductions
__global__ void __launch_bounds__(256) k_ln(
    const float* __restrict__ in, const float* __restrict__ g,
    const float* __restrict__ b, float* __restrict__ out) {
  const int tid = threadIdx.x, lane = tid & 31, wave = tid >> 5;
  const size_t row = (size_t)blockIdx.x * 8 + wave;
  const float* p = in + row * H_LEN;
  float v[8], s1 = 0.f, s2 = 0.f;
#pragma unroll
  for (int i = 0; i < 8; ++i) {
    v[i] = p[lane + 32 * i];
    s1 += v[i];
    s2 += v[i] * v[i];
  }
#pragma unroll
  for (int off = 16; off >= 1; off >>= 1) {
    s1 += __shfl_xor(s1, off, 32);
    s2 += __shfl_xor(s2, off, 32);
  }
  const float mean = s1 * (1.f / H_LEN);
  const float rstd = rsqrtf(s2 * (1.f / H_LEN) - mean * mean + EPSV);
  float* q = out + row * H_LEN;
#pragma unroll
  for (int i = 0; i < 8; ++i) {
    const int c = lane + 32 * i;
    q[c] = (v[i] - mean) * rstd * g[c] + b[c];
  }
}

// head: relu(hx@nn1.T+b1) -> LN(8) -> @nn2.T+b2 ; wave32 per row
__global__ void __launch_bounds__(256) k_head(
    const float* __restrict__ in, const float* __restrict__ w1,
    const float* __restrict__ b1, const float* __restrict__ lg,
    const float* __restrict__ lb, const float* __restrict__ w2,
    const float* __restrict__ b2, float* __restrict__ out) {
  const int tid = threadIdx.x, lane = tid & 31, wave = tid >> 5;
  const size_t row = (size_t)blockIdx.x * 8 + wave;
  const int o = lane & 7, seg = lane >> 3;
  const float* p = in + row * H_LEN;
  const float* wr = w1 + o * H_LEN;
  float acc = 0.f;
  for (int k = seg * 64; k < seg * 64 + 64; ++k) acc += p[k] * wr[k];
  acc += __shfl_xor(acc, 8, 32);
  acc += __shfl_xor(acc, 16, 32);
  const float y = fmaxf(acc + b1[o], 0.f);
  float s1 = y, s2 = y * y;
#pragma unroll
  for (int off = 4; off >= 1; off >>= 1) {
    s1 += __shfl_xor(s1, off, 32);
    s2 += __shfl_xor(s2, off, 32);
  }
  const float mean = s1 * 0.125f;
  const float rstd = rsqrtf(s2 * 0.125f - mean * mean + EPSV);
  const float yn = (y - mean) * rstd * lg[o] + lb[o];
  float t0 = yn * w2[o], t1 = yn * w2[8 + o];
#pragma unroll
  for (int off = 4; off >= 1; off >>= 1) {
    t0 += __shfl_xor(t0, off, 32);
    t1 += __shfl_xor(t1, off, 32);
  }
  if (lane == 0) {
    out[row * 2 + 0] = t0 + b2[0];
    out[row * 2 + 1] = t1 + b2[1];
  }
}

// ---------------------------------------------------------------------------
extern "C" void kernel_launch(void* const* d_in, const int* in_sizes, int n_in,
                              void* d_out, int out_size, void* d_ws, size_t ws_size,
                              hipStream_t stream) {
  (void)in_sizes; (void)n_in; (void)out_size; (void)ws_size;
  const float* X       = (const float*)d_in[0];
  const float* M       = (const float*)d_in[1];
  const float* mask    = (const float*)d_in[2];
  const float* i2h_w   = (const float*)d_in[3];
  const float* i2h_b   = (const float*)d_in[4];
  const float* comb_w  = (const float*)d_in[5];
  const float* comb_b  = (const float*)d_in[6];
  const float* down_w  = (const float*)d_in[7];
  const float* down_b  = (const float*)d_in[8];
  const float* conv1_w = (const float*)d_in[9];
  const float* conv1_b = (const float*)d_in[10];
  const float* conv2_w = (const float*)d_in[11];
  const float* conv2_b = (const float*)d_in[12];
  const float* bn1_g   = (const float*)d_in[13];
  const float* bn1_b   = (const float*)d_in[14];
  const float* bn2_g   = (const float*)d_in[15];
  const float* bn2_b   = (const float*)d_in[16];
  const float* r1_wih  = (const float*)d_in[17];
  const float* r1_whh  = (const float*)d_in[18];
  const float* r1_bih  = (const float*)d_in[19];
  const float* r1_bhh  = (const float*)d_in[20];
  const float* r1_cw   = (const float*)d_in[21];
  const float* r1_cb   = (const float*)d_in[22];
  const float* r2_wih  = (const float*)d_in[23];
  const float* r2_whh  = (const float*)d_in[24];
  const float* r2_bih  = (const float*)d_in[25];
  const float* r2_bhh  = (const float*)d_in[26];
  const float* r2_cw   = (const float*)d_in[27];
  const float* r2_cb   = (const float*)d_in[28];
  const float* ln1_g   = (const float*)d_in[29];
  const float* ln1_b   = (const float*)d_in[30];
  const float* ln2_g   = (const float*)d_in[31];
  const float* ln2_b   = (const float*)d_in[32];
  const float* ln3_g   = (const float*)d_in[33];
  const float* ln3_b   = (const float*)d_in[34];
  const float* nn1_w   = (const float*)d_in[35];
  const float* nn1_b   = (const float*)d_in[36];
  const float* nn2_w   = (const float*)d_in[37];
  const float* nn2_b   = (const float*)d_in[38];

  // ---- workspace carve: f16 weights | stats | h0 | 3 big f32 bufs | BN partials
  h16* conv2f = (h16*)d_ws;                  // 256*256
  h16* wih1 = conv2f + 256 * 256;            // 768*256
  h16* whh1 = wih1 + 768 * 256;
  h16* cw1  = whh1 + 768 * 256;              // 256*512
  h16* wih2 = cw1 + 256 * 512;
  h16* whh2 = wih2 + 768 * 256;
  h16* cw2  = whh2 + 768 * 256;
  float* fb = (float*)(cw2 + 256 * 512);
  float* sc1 = fb;          float* sh1 = fb + 256;
  float* sc2 = fb + 512;    float* sh2 = fb + 768;
  float* h0f  = fb + 1024;                             // B*H
  float* bigA = h0f + (size_t)B_LEN * H_LEN;           // NROWS*H each (128 MB)
  float* bigB = bigA + (size_t)NROWS * H_LEN;
  float* bigC = bigB + (size_t)NROWS * H_LEN;
  float* part1 = bigC + (size_t)NROWS * H_LEN;         // C1_BLOCKS*512 (8 MB)
  float* part2 = part1 + (size_t)C1_BLOCKS * 512;      // C2_BLOCKS*512 (4 MB)

  // one-shot f32->f16 weight conversion into workspace
  k_cvt_f32_f16<<<128, 256, 0, stream>>>(conv2_w, conv2f, 256 * 256);
  k_cvt_f32_f16<<<256, 256, 0, stream>>>(r1_wih, wih1, 768 * 256);
  k_cvt_f32_f16<<<256, 256, 0, stream>>>(r1_whh, whh1, 768 * 256);
  k_cvt_f32_f16<<<256, 256, 0, stream>>>(r1_cw,  cw1,  256 * 512);
  k_cvt_f32_f16<<<256, 256, 0, stream>>>(r2_wih, wih2, 768 * 256);
  k_cvt_f32_f16<<<256, 256, 0, stream>>>(r2_whh, whh2, 768 * 256);
  k_cvt_f32_f16<<<256, 256, 0, stream>>>(r2_cw,  cw2,  256 * 512);

  k_h0<<<B_LEN, H_LEN, 0, stream>>>(X, i2h_w, i2h_b, h0f);

  k_conv1<<<C1_BLOCKS, 256, 0, stream>>>(X, M, comb_w, comb_b, conv1_w, conv1_b,
                                         bigA, part1);
  k_bnredfin<<<1, 256, 0, stream>>>(part1, C1_BLOCKS, bn1_g, bn1_b, sc1, sh1);
  k_conv2<<<C2_BLOCKS, 128, 0, stream>>>(bigA, sc1, sh1, conv2f, conv2_b, bigB,
                                         part2);
  k_bnredfin<<<1, 256, 0, stream>>>(part2, C2_BLOCKS, bn2_g, bn2_b, sc2, sh2);
  k_hx<<<NROWS, 256, 0, stream>>>(X, bigB, sc2, sh2, down_w, down_b, bigC);

  k_gru<<<16, 256, 0, stream>>>(bigC, h0f, mask, wih1, whh1, cw1, r1_bih, r1_bhh,
                                r1_cb, bigA);
  k_ln<<<NROWS / 8, 256, 0, stream>>>(bigA, ln1_g, ln1_b, bigB);
  k_gru<<<16, 256, 0, stream>>>(bigB, h0f, mask, wih2, whh2, cw2, r2_bih, r2_bhh,
                                r2_cb, bigC);
  k_ln<<<NROWS / 8, 256, 0, stream>>>(bigC, ln2_g, ln2_b, bigA);

  k_head<<<NROWS / 8, 256, 0, stream>>>(bigA, nn1_w, nn1_b, ln3_g, ln3_b, nn2_w,
                                        nn2_b, (float*)d_out);
}
